// PatchNet_40999757807796
// MI455X (gfx1250) — matrix operand order repaired and checked by reading, not compiled, not run on hardware
//
#include <hip/hip_runtime.h>
#include <math.h>

typedef __attribute__((ext_vector_type(16))) _Float16 v16h;
typedef __attribute__((ext_vector_type(8)))  _Float16 h8;
typedef __attribute__((ext_vector_type(2)))  _Float16 h2;
typedef __attribute__((ext_vector_type(8)))  float    v8f;

#define EPSV 1e-5f

// ---------------------------------------------------------------------------
// Tiled fused GEMM  Y[b,o,s] = epilogue( sum_c W[o,c]*X[b,c,s] + bias[o] )
// Wh[Opad][Kp] f16, zero-padded (Opad mult of O-tile, Kp mult 32).
// 256 threads = 8 waves; OT(O) x 64(S) macro-tile (OT = 64 or 128); K-step 32;
// double-buffered LDS, one barrier per K-step, next tile in regs during WMMA.
// All fragments loaded before the WMMA chain so ds_loads clause together and
// the scheduler can use partial s_wait_dscnt.
// mode 0: bias  1: relu(x+b)  2: bn(relu(x+b))  3: relu(bn(x+b))
// ---------------------------------------------------------------------------
template <int OT>
__global__ __launch_bounds__(256) void k_gemm_wmma(
    const _Float16* __restrict__ Wh, const float* __restrict__ X, float* __restrict__ Y,
    const float* __restrict__ bias,
    const float* __restrict__ bng, const float* __restrict__ bnb,
    const float* __restrict__ bnm, const float* __restrict__ bnv,
    int C, int Kp, int O, int S, long xBat, long yBat, int mode)
{
  constexpr int NT = (OT == 128) ? 4 : 2;    // WMMA tiles per wave
  __shared__ _Float16 As[2][OT * 40];        // [row(o)][k 32] padded to 40
  __shared__ _Float16 Bs[2][64 * 40];        // [col(s)][k 32] padded (transposed)
  const int tid  = threadIdx.x;
  const int wid  = tid >> 5;
  const int lane = tid & 31;
  const int half = lane >> 4;
  const int mr   = lane & 15;
  const int sTile = blockIdx.x * 64;
  const int oTile = blockIdx.y * OT;
  const int bat   = blockIdx.z;
  const float* Xb = X + (long)bat * xBat;
  const int oSub = (OT == 128) ? wid * 16 : (wid & 3) * 16;
  const int sSub = (OT == 128) ? 0 : (wid >> 2) * 32;
  const bool edgeS = (sTile + 64 > S);

  // ---- staging assignments ----
  int arow, akoff;
  if (OT == 128) { arow = tid >> 1; akoff = (tid & 1) * 16; }
  else           { arow = tid >> 2; akoff = (tid & 3) * 8;  }
  const int bk2 = (tid >> 4) * 2;      // k pair: 0,2,..,30
  const int bs4 = (tid & 15) * 4;      // col group: 0,4,..,60

  h8    av0, av1;     // A regs (b128-aligned)
  float r0[4], r1[4]; // B regs: rows bk2, bk2+1 x 4 cols

  auto loadA = [&](int kk) {
    const _Float16* p = &Wh[(long)(oTile + arow) * Kp + kk + akoff];
    av0 = *(const h8*)p;
    if (OT == 128) av1 = *(const h8*)(p + 8);
  };
  auto storeA = [&](int buf) {
    _Float16* p = &As[buf][arow * 40 + akoff];
    *(h8*)p = av0;
    if (OT == 128) *(h8*)(p + 8) = av1;
  };
  auto loadB = [&](int kk) {
    if (!edgeS && kk + 32 <= C) {
      float4 v0 = *(const float4*)&Xb[(long)(kk + bk2) * S + sTile + bs4];
      float4 v1 = *(const float4*)&Xb[(long)(kk + bk2 + 1) * S + sTile + bs4];
      r0[0] = v0.x; r0[1] = v0.y; r0[2] = v0.z; r0[3] = v0.w;
      r1[0] = v1.x; r1[1] = v1.y; r1[2] = v1.z; r1[3] = v1.w;
    } else {
      int ck0 = kk + bk2;     ck0 = ck0 < C ? ck0 : C - 1;  // A is 0 beyond C
      int ck1 = kk + bk2 + 1; ck1 = ck1 < C ? ck1 : C - 1;
#pragma unroll
      for (int j = 0; j < 4; ++j) {
        int cs = sTile + bs4 + j; cs = cs < S ? cs : S - 1; // col never stored
        r0[j] = Xb[(long)ck0 * S + cs];
        r1[j] = Xb[(long)ck1 * S + cs];
      }
    }
  };
  auto storeB = [&](int buf) {
#pragma unroll
    for (int j = 0; j < 4; ++j) {
      h2 pk; pk[0] = (_Float16)r0[j]; pk[1] = (_Float16)r1[j];
      *(h2*)&Bs[buf][(bs4 + j) * 40 + bk2] = pk;   // packed b32 store
    }
  };

  v8f acc[NT];
#pragma unroll
  for (int t = 0; t < NT; ++t) acc[t] = (v8f){};

  loadA(0); loadB(0);
  storeA(0); storeB(0);
  int cur = 0;
  for (int kk = 0; kk < Kp; kk += 32) {
    const bool hasNext = (kk + 32) < Kp;       // uniform
    if (hasNext) { loadA(kk + 32); loadB(kk + 32); }
    __syncthreads();
    // ---- all fragments first (clause the ds_loads), then the WMMA chain ----
    const int ar = (oSub + mr) * 40;
    h8 a0 = *(const h8*)&As[cur][ar + half * 8];
    h8 a1 = *(const h8*)&As[cur][ar + 16 + half * 8];
    v16h a = __builtin_shufflevector(a0, a1, 0, 1, 2, 3, 4, 5, 6, 7,
                                             8, 9, 10, 11, 12, 13, 14, 15);
    v16h bf[NT];
#pragma unroll
    for (int t = 0; t < NT; ++t) {
      const int br = (sSub + t * 16 + mr) * 40 + half * 8;
      h8 bl = *(const h8*)&Bs[cur][br];
      h8 bh = *(const h8*)&Bs[cur][br + 16];
      bf[t] = __builtin_shufflevector(bl, bh, 0, 1, 2, 3, 4, 5, 6, 7,
                                              8, 9, 10, 11, 12, 13, 14, 15);
    }
#pragma unroll
    for (int t = 0; t < NT; ++t)
      acc[t] = __builtin_amdgcn_wmma_f32_16x16x32_f16(false, a, false, bf[t],
                                                      (short)0, acc[t], false, false);
    if (hasNext) { storeA(cur ^ 1); storeB(cur ^ 1); cur ^= 1; }
  }

  // ---- epilogue ----
  float* Yb = Y + (long)bat * yBat;
#pragma unroll
  for (int t = 0; t < NT; ++t) {
    int col = sTile + sSub + t * 16 + mr;
    if (col >= S) continue;
#pragma unroll
    for (int i = 0; i < 8; ++i) {
      int row = oTile + oSub + half * 8 + i;   // D: VGPR i -> M = i + 8*half
      if (row >= O) continue;
      float v = acc[t][i] + (bias ? bias[row] : 0.0f);
      if (mode == 1) {
        v = fmaxf(v, 0.0f);
      } else if (mode == 2) {                  // bn(relu(x))
        v = fmaxf(v, 0.0f);
        float sc = bng[row] * rsqrtf(bnv[row] + EPSV);
        v = (v - bnm[row]) * sc + bnb[row];
      } else if (mode == 3) {                  // relu(bn(x))
        float sc = bng[row] * rsqrtf(bnv[row] + EPSV);
        v = fmaxf((v - bnm[row]) * sc + bnb[row], 0.0f);
      }
      Yb[(long)row * S + col] = v;
    }
  }
}

// Weight conversion: Wh[Opad][Kp] f16, zero-padded.
__global__ void k_convert_w(const float* __restrict__ W, _Float16* __restrict__ Wh,
                            int O, int C, int Kp, long total)
{
  long t = (long)blockIdx.x * blockDim.x + threadIdx.x;
  if (t >= total) return;
  int k = t % Kp; long o = t / Kp;
  float f = 0.0f;
  if (o < O && k < C) f = W[o * C + k];
  Wh[t] = (_Float16)f;
}

// ---------------------------------------------------------------------------
// Farthest point sampling: one block per batch, iterative argmax.
// ---------------------------------------------------------------------------
__global__ void k_fps(const float* __restrict__ xyz, int N, int npoint,
                      int* __restrict__ sidx, float* __restrict__ dist)
{
  const int bat = blockIdx.x;
  const float* P = xyz + (long)bat * N * 3;
  float* D = dist + (long)bat * N;
  __shared__ float sv[256];
  __shared__ int   si[256];
  __shared__ int   sfar;
  const int tid = threadIdx.x;
  for (int i = tid; i < N; i += blockDim.x) D[i] = 1e10f;
  if (tid == 0) sfar = 0;
  __syncthreads();
  for (int it = 0; it < npoint; ++it) {
    int far = sfar;
    if (tid == 0) sidx[bat * npoint + it] = far;
    float cx = P[far * 3 + 0], cy = P[far * 3 + 1], cz = P[far * 3 + 2];
    float best = -1.0f; int bi = 0x7fffffff;
    for (int i = tid; i < N; i += blockDim.x) {
      float dx = P[i * 3 + 0] - cx, dy = P[i * 3 + 1] - cy, dz = P[i * 3 + 2] - cz;
      float d = dx * dx + dy * dy + dz * dz;
      float nd = fminf(D[i], d);
      D[i] = nd;
      if (nd > best) { best = nd; bi = i; }
    }
    sv[tid] = best; si[tid] = bi;
    __syncthreads();
    for (int s = blockDim.x / 2; s > 0; s >>= 1) {
      if (tid < s) {
        if (sv[tid + s] > sv[tid] ||
            (sv[tid + s] == sv[tid] && si[tid + s] < si[tid])) {
          sv[tid] = sv[tid + s]; si[tid] = si[tid + s];
        }
      }
      __syncthreads();
    }
    if (tid == 0) sfar = si[0];
    __syncthreads();
  }
}

__global__ void k_gather_xyz(const float* __restrict__ xyz, const int* __restrict__ sidx,
                             float* __restrict__ out, int nsrc, int S, long total)
{
  long t = (long)blockIdx.x * blockDim.x + threadIdx.x;
  if (t >= total) return;
  int c = t % 3; long r = t / 3;
  int s = r % S; int b = r / S;
  out[t] = xyz[((long)b * nsrc + sidx[b * S + s]) * 3 + c];
}

// Ball query: first K indices (ascending) with d^2 <= r^2; pad with first hit.
__global__ void k_ballquery(const float* __restrict__ xyz, const float* __restrict__ nxyz,
                            int N, int S, int K, float r2, int* __restrict__ gi)
{
  int bat = blockIdx.y;
  int s = blockIdx.x * blockDim.x + threadIdx.x;
  if (s >= S) return;
  const float* P = xyz + (long)bat * N * 3;
  const float* q = nxyz + ((long)bat * S + s) * 3;
  int* out = gi + ((long)bat * S + s) * K;
  int cnt = 0, first = 0;
  for (int i = 0; i < N && cnt < K; ++i) {
    float dx = P[i * 3 + 0] - q[0], dy = P[i * 3 + 1] - q[1], dz = P[i * 3 + 2] - q[2];
    if (dx * dx + dy * dy + dz * dz <= r2) {
      if (cnt == 0) first = i;
      out[cnt++] = i;
    }
  }
  for (int k = cnt; k < K; ++k) out[k] = first;
}

// Grouped input: out[b][c][s*K+k]; feats channels first, then (xyz - center).
__global__ void k_build_grouped(const float* __restrict__ feats, const float* __restrict__ xyz,
                                const float* __restrict__ nxyz, const int* __restrict__ gi,
                                float* __restrict__ out, int Cf, int nsrc, int S, int K, long total)
{
  long t = (long)blockIdx.x * blockDim.x + threadIdx.x;
  if (t >= total) return;
  int k = t % K; long r = t / K;
  int s = r % S; r /= S;
  int c = r % (Cf + 3); int b = r / (Cf + 3);
  int src = gi[((long)b * S + s) * K + k];
  float v;
  if (c < Cf) v = feats[((long)b * Cf + c) * nsrc + src];
  else {
    int cc = c - Cf;
    v = xyz[((long)b * nsrc + src) * 3 + cc] - nxyz[((long)b * S + s) * 3 + cc];
  }
  out[t] = v;
}

__global__ void k_maxpool_k(const float* __restrict__ in, float* __restrict__ out,
                            int O, int S, int K, long inBat, long outBat, long total)
{
  long t = (long)blockIdx.x * blockDim.x + threadIdx.x;
  if (t >= total) return;
  int s = t % S; long r = t / S;
  int o = r % O; int b = r / O;
  const float* p = in + (long)b * inBat + (long)o * S * K + (long)s * K;
  float m = -INFINITY;
  for (int k = 0; k < K; ++k) m = fmaxf(m, p[k]);
  out[(long)b * outBat + (long)o * S + s] = m;
}

__global__ void k_maxpool_s(const float* __restrict__ in, float* __restrict__ out,
                            int O, int S, long inBat, long total)
{
  long t = (long)blockIdx.x * blockDim.x + threadIdx.x;
  if (t >= total) return;
  int o = t % O; int b = t / O;
  const float* p = in + (long)b * inBat + (long)o * S;
  float m = -INFINITY;
  for (int s = 0; s < S; ++s) m = fmaxf(m, p[s]);
  out[(long)b * O + o] = m;
}

// [bat][n][c] -> [bat][c][n]
__global__ void k_nc_to_cn(const float* __restrict__ in, float* __restrict__ out,
                           int n, int c, long total)
{
  long t = (long)blockIdx.x * blockDim.x + threadIdx.x;
  if (t >= total) return;
  int ni = t % n; long r = t / n;
  int ci = r % c; int b = r / c;
  out[t] = in[((long)b * n + ni) * c + ci];
}

// sa3 input: channels [xyz(3) | feats(Cf)] over S positions
__global__ void k_concat_xyz_feats(const float* __restrict__ xyz, const float* __restrict__ feats,
                                   float* __restrict__ out, int Cf, int S, long total)
{
  long t = (long)blockIdx.x * blockDim.x + threadIdx.x;
  if (t >= total) return;
  int s = t % S; long r = t / S;
  int c = r % (Cf + 3); int b = r / (Cf + 3);
  out[t] = (c < 3) ? xyz[((long)b * S + s) * 3 + c]
                   : feats[((long)b * Cf + (c - 3)) * S + s];
}

// px[(p*B+b)][m][c] = x[b][patch_idx[p][m]][c]
__global__ void k_build_px(const float* __restrict__ x, const int* __restrict__ pidx,
                           float* __restrict__ px, int Bn, int Pn, int N, int M, long total)
{
  long t = (long)blockIdx.x * blockDim.x + threadIdx.x;
  if (t >= total) return;
  int c = t % 3; long r = t / 3;
  int m = r % M; r /= M;
  int b = r % Bn; int p = r / Bn;
  px[(((long)p * Bn + b) * M + m) * 3 + c] =
      x[((long)b * N + pidx[p * M + m]) * 3 + c];
}

// rx[b][c4096][n64] via raw reshape of (B,P,P,2C) -> (B, 2C, P*P)
__global__ void k_build_rx(const float* __restrict__ g, const float* __restrict__ pf,
                           float* __restrict__ rx, long total)
{
  long t = (long)blockIdx.x * blockDim.x + threadIdx.x;
  if (t >= total) return;
  int n = t % 64; long r = t / 64;
  int c = r % 4096; int b = r / 4096;
  long flat = (long)c * 64 + n;          // index into (P=8, P=8, 4096)
  int p1 = (int)(flat >> 15);            // / (8*4096)
  long rem = flat & 32767;
  int p2 = (int)(rem >> 12);             // / 4096
  int cc = (int)(rem & 4095);
  int p, ch;
  if (cc < 2048) { p = p2; ch = cc; } else { p = p1; ch = cc - 2048; }
  float v = (ch < 1024) ? g[(long)b * 1024 + ch]
                        : pf[((long)p * 8 + b) * 1024 + (ch - 1024)];
  rx[t] = v;
}

__global__ void k_fc(const float* __restrict__ in, const float* __restrict__ W,
                     const float* __restrict__ bias, float* __restrict__ out,
                     int Bn, int C, int O, int relu)
{
  int t = blockIdx.x * blockDim.x + threadIdx.x;
  if (t >= Bn * O) return;
  int o = t % O; int b = t / O;
  float s = bias[o];
  for (int c = 0; c < C; ++c) s += in[(long)b * C + c] * W[(long)o * C + c];
  if (relu) s = fmaxf(s, 0.0f);
  out[(long)b * O + o] = s;
}

// ---------------------------------------------------------------------------
// Host-side orchestration
// ---------------------------------------------------------------------------
struct BNp   { const float *g, *b, *m, *v; };
struct ConvP { const float *w, *b; };
struct LayerP { ConvP c; BNp bn; };
struct ConvH { const _Float16* w; int Kp; };

struct Net {
  ConvP conv1, conv2;
  BNp   bn1, bn2;
  LayerP sa1[3][3], sa2[3][3], sa3[3];
  ConvP rl_r[3], rl_axis[3], rl_t[3];
  ConvP fc_r[2], fc_axis[2], fc_t[2];
  const int*   patch_idx;
  const float* x;
};

struct NetH {
  ConvH conv1, conv2;
  ConvH sa1[3][3], sa2[3][3], sa3[3];
  ConvH rl_r[3], rl_axis[3], rl_t[3];
};

struct WS {
  float *px_all, *xc, *f0, *f1;
  float *l1xyz, *l1f, *l2xyz, *l2f, *sa3in;
  float *hin, *hA, *hB, *dist;
  float *gfeat, *pf, *rx, *vecA, *vecB;
  int   *sidx, *gi;
};

static const int   SA1_CH[3][3] = {{64,64,64},{64,64,128},{64,96,128}};
static const float SA1_R[3] = {0.01f, 0.02f, 0.04f};
static const int   SA1_K[3] = {16, 32, 128};
static const int   SA2_CH[3][3] = {{64,64,128},{128,128,256},{128,128,256}};
static const float SA2_R[3] = {0.02f, 0.04f, 0.08f};
static const int   SA2_K[3] = {32, 64, 128};
static const int   SA3_CH[3] = {256, 512, 1024};
static const int   RL_CH[3]  = {2048, 1024, 512};

static inline long cdiv(long a, long b) { return (a + b - 1) / b; }

static void gemm(hipStream_t st, const ConvH& Wh, const float* X, float* Y,
                 const float* bias, const BNp* bn,
                 int C, int O, int S, int bat, int mode, long xB, long yB)
{
  const float* g_ = bn ? bn->g : nullptr;
  const float* b_ = bn ? bn->b : nullptr;
  const float* m_ = bn ? bn->m : nullptr;
  const float* v_ = bn ? bn->v : nullptr;
  if (O % 128 == 0) {
    dim3 g((unsigned)cdiv(S, 64), (unsigned)(O / 128), (unsigned)bat);
    k_gemm_wmma<128><<<g, 256, 0, st>>>(Wh.w, X, Y, bias, g_, b_, m_, v_,
                                        C, Wh.Kp, O, S, xB, yB, mode);
  } else {
    dim3 g((unsigned)cdiv(S, 64), (unsigned)cdiv(O, 64), (unsigned)bat);
    k_gemm_wmma<64><<<g, 256, 0, st>>>(Wh.w, X, Y, bias, g_, b_, m_, v_,
                                       C, Wh.Kp, O, S, xB, yB, mode);
  }
}

static void sa_stage(hipStream_t st, const float* xyz, const float* feats,
                     int bat, int nsrc, int Cf, int S,
                     const float* radii, const int* Ks,
                     const LayerP (&br)[3][3], const ConvH (&brh)[3][3],
                     const int (&ch)[3][3],
                     float* out_xyz, float* out_f, int Ctot, WS& w)
{
  k_fps<<<bat, 256, 0, st>>>(xyz, nsrc, S, w.sidx, w.dist);
  {
    long tot = (long)bat * S * 3;
    k_gather_xyz<<<(unsigned)cdiv(tot, 256), 256, 0, st>>>(xyz, w.sidx, out_xyz, nsrc, S, tot);
  }
  int c0 = 0;
  for (int b = 0; b < 3; ++b) {
    int K = Ks[b];
    dim3 bg((unsigned)cdiv(S, 64), (unsigned)bat);
    k_ballquery<<<bg, 64, 0, st>>>(xyz, out_xyz, nsrc, S, K, radii[b] * radii[b], w.gi);
    int Cin = Cf + 3;
    long tot = (long)bat * Cin * S * K;
    k_build_grouped<<<(unsigned)cdiv(tot, 256), 256, 0, st>>>(
        feats, xyz, out_xyz, w.gi, w.hin, Cf, nsrc, S, K, tot);
    int SK = S * K;
    const float* cur = w.hin; int curC = Cin;
    float* bufs[3] = { w.hA, w.hB, w.hA };
    for (int l = 0; l < 3; ++l) {
      gemm(st, brh[b][l], cur, bufs[l], br[b][l].c.b, &br[b][l].bn,
           curC, ch[b][l], SK, bat, 3, (long)curC * SK, (long)ch[b][l] * SK);
      cur = bufs[l]; curC = ch[b][l];
    }
    long tot2 = (long)bat * curC * S;
    k_maxpool_k<<<(unsigned)cdiv(tot2, 256), 256, 0, st>>>(
        cur, out_f + (long)c0 * S, curC, S, K, (long)curC * SK, (long)Ctot * S, tot2);
    c0 += curC;
  }
}

static void patch_feat_run(hipStream_t st, const Net& net, const NetH& nh,
                           const float* pts, int bat, int n, float* outfeat, WS& w)
{
  long tot = (long)bat * n * 3;
  k_nc_to_cn<<<(unsigned)cdiv(tot, 256), 256, 0, st>>>(pts, w.xc, n, 3, tot);
  gemm(st, nh.conv1, w.xc, w.f0, net.conv1.b, &net.bn1, 3, 64, n, bat, 2, 3L * n, 64L * n);
  gemm(st, nh.conv2, w.f0, w.f1, net.conv2.b, &net.bn2, 64, 128, n, bat, 2, 64L * n, 128L * n);
  sa_stage(st, pts, w.f1, bat, n, 128, 128, SA1_R, SA1_K, net.sa1, nh.sa1, SA1_CH,
           w.l1xyz, w.l1f, 320, w);
  sa_stage(st, w.l1xyz, w.l1f, bat, 128, 320, 64, SA2_R, SA2_K, net.sa2, nh.sa2, SA2_CH,
           w.l2xyz, w.l2f, 640, w);
  long t3 = (long)bat * 643 * 64;
  k_concat_xyz_feats<<<(unsigned)cdiv(t3, 256), 256, 0, st>>>(w.l2xyz, w.l2f, w.sa3in, 640, 64, t3);
  const float* cur = w.sa3in; int curC = 643;
  float* bufs[3] = { w.hA, w.hB, w.hA };
  for (int l = 0; l < 3; ++l) {
    gemm(st, nh.sa3[l], cur, bufs[l], net.sa3[l].c.b, &net.sa3[l].bn,
         curC, SA3_CH[l], 64, bat, 3, (long)curC * 64, (long)SA3_CH[l] * 64);
    cur = bufs[l]; curC = SA3_CH[l];
  }
  long t4 = (long)bat * 1024;
  k_maxpool_s<<<(unsigned)cdiv(t4, 256), 256, 0, st>>>(cur, outfeat, 1024, 64, 1024L * 64, t4);
}

static void run_head(hipStream_t st, const ConvP rl[3], const ConvH rlh[3],
                     const ConvP fc[2], const float* rx, float* outSeg,
                     int fcMid, int fcOut, WS& w)
{
  const float* cur = rx; int curC = 4096;
  float* bufs[3] = { w.hA, w.hB, w.hA };
  for (int l = 0; l < 3; ++l) {
    gemm(st, rlh[l], cur, bufs[l], rl[l].b, nullptr,
         curC, RL_CH[l], 64, 8, 1, (long)curC * 64, (long)RL_CH[l] * 64);
    cur = bufs[l]; curC = RL_CH[l];
  }
  long t = 8L * 512;
  k_maxpool_s<<<(unsigned)cdiv(t, 256), 256, 0, st>>>(cur, w.vecA, 512, 64, 512L * 64, t);
  k_fc<<<(unsigned)cdiv(8L * fcMid, 256), 256, 0, st>>>(w.vecA, fc[0].w, fc[0].b, w.vecB, 8, 512, fcMid, 1);
  k_fc<<<(unsigned)cdiv(8L * fcOut, 256), 256, 0, st>>>(w.vecB, fc[1].w, fc[1].b, outSeg, 8, fcMid, fcOut, 0);
}

extern "C" void kernel_launch(void* const* d_in, const int* in_sizes, int n_in,
                              void* d_out, int out_size, void* d_ws, size_t ws_size,
                              hipStream_t stream)
{
  (void)in_sizes; (void)n_in; (void)out_size; (void)ws_size;
  // --- parse params (JAX pytree order: dict keys sorted alphabetically) ---
  Net net;
  int ii = 0;
  auto nf = [&]() { return (const float*)d_in[ii++]; };
  auto getBN   = [&](BNp& p)   { p.b = nf(); p.g = nf(); p.m = nf(); p.v = nf(); };
  auto getConv = [&](ConvP& p) { p.b = nf(); p.w = nf(); };
  auto getLayer = [&](LayerP& L) { getBN(L.bn); getConv(L.c); };
  getBN(net.bn1); getBN(net.bn2);
  getConv(net.conv1); getConv(net.conv2);
  for (int i = 0; i < 2; ++i) getConv(net.fc_axis[i]);
  for (int i = 0; i < 2; ++i) getConv(net.fc_r[i]);
  for (int i = 0; i < 2; ++i) getConv(net.fc_t[i]);
  for (int i = 0; i < 3; ++i) getConv(net.rl_axis[i]);
  for (int i = 0; i < 3; ++i) getConv(net.rl_r[i]);
  for (int i = 0; i < 3; ++i) getConv(net.rl_t[i]);
  for (int b = 0; b < 3; ++b) for (int l = 0; l < 3; ++l) getLayer(net.sa1[b][l]);
  for (int b = 0; b < 3; ++b) for (int l = 0; l < 3; ++l) getLayer(net.sa2[b][l]);
  for (int l = 0; l < 3; ++l) getLayer(net.sa3[l]);
  net.patch_idx = (const int*)d_in[ii++];
  net.x = (const float*)d_in[ii++];

  // --- workspace bump allocator ---
  char* wp = (char*)d_ws;
  auto alloc = [&](size_t bytes) {
    void* p = (void*)wp;
    wp += (bytes + 255) & ~(size_t)255;
    return p;
  };
  WS w;
  w.px_all = (float*)alloc(sizeof(float) * 64 * 512 * 3);
  w.xc     = (float*)alloc(sizeof(float) * 8 * 3 * 2000);
  w.f0     = (float*)alloc(sizeof(float) * 8 * 64 * 2000);
  w.f1     = (float*)alloc(sizeof(float) * 8 * 128 * 2000);
  w.l1xyz  = (float*)alloc(sizeof(float) * 8 * 128 * 3);
  w.l1f    = (float*)alloc(sizeof(float) * 8 * 320 * 128);
  w.l2xyz  = (float*)alloc(sizeof(float) * 8 * 64 * 3);
  w.l2f    = (float*)alloc(sizeof(float) * 8 * 640 * 64);
  w.sa3in  = (float*)alloc(sizeof(float) * 8 * 643 * 64);
  w.hin    = (float*)alloc(sizeof(float) * 8L * 323 * 64 * 128);   // max grouped input
  w.hA     = (float*)alloc(sizeof(float) * 8L * 128 * 128 * 128);  // max layer output
  w.hB     = (float*)alloc(sizeof(float) * 8L * 128 * 128 * 128);
  w.dist   = (float*)alloc(sizeof(float) * 8 * 2000);
  w.gfeat  = (float*)alloc(sizeof(float) * 8 * 1024);
  w.pf     = (float*)alloc(sizeof(float) * 64 * 1024);
  w.rx     = (float*)alloc(sizeof(float) * 8L * 4096 * 64);
  w.vecA   = (float*)alloc(sizeof(float) * 8 * 512);
  w.vecB   = (float*)alloc(sizeof(float) * 8 * 512);
  w.sidx   = (int*)alloc(sizeof(int) * 8 * 128);
  w.gi     = (int*)alloc(sizeof(int) * 8 * 128 * 128);

  // --- pre-convert all GEMM weights to padded f16 (once per launch) ---
  auto convW = [&](const ConvP& c, int O, int C) {
    int Kp = (C + 31) & ~31;
    long Op = ((long)O + 127) & ~127L;   // covers both 64/128 tile variants
    _Float16* p = (_Float16*)alloc(sizeof(_Float16) * (size_t)Op * Kp);
    long tot = Op * Kp;
    k_convert_w<<<(unsigned)cdiv(tot, 256), 256, 0, stream>>>(c.w, p, O, C, Kp, tot);
    ConvH h; h.w = p; h.Kp = Kp; return h;
  };
  NetH nh;
  nh.conv1 = convW(net.conv1, 64, 3);
  nh.conv2 = convW(net.conv2, 128, 64);
  for (int b = 0; b < 3; ++b)
    for (int l = 0; l < 3; ++l) {
      int C = (l == 0) ? 131 : SA1_CH[b][l - 1];
      nh.sa1[b][l] = convW(net.sa1[b][l].c, SA1_CH[b][l], C);
    }
  for (int b = 0; b < 3; ++b)
    for (int l = 0; l < 3; ++l) {
      int C = (l == 0) ? 323 : SA2_CH[b][l - 1];
      nh.sa2[b][l] = convW(net.sa2[b][l].c, SA2_CH[b][l], C);
    }
  for (int l = 0; l < 3; ++l) {
    int C = (l == 0) ? 643 : SA3_CH[l - 1];
    nh.sa3[l] = convW(net.sa3[l].c, SA3_CH[l], C);
  }
  for (int l = 0; l < 3; ++l) {
    int C = (l == 0) ? 4096 : RL_CH[l - 1];
    nh.rl_r[l]    = convW(net.rl_r[l],    RL_CH[l], C);
    nh.rl_axis[l] = convW(net.rl_axis[l], RL_CH[l], C);
    nh.rl_t[l]    = convW(net.rl_t[l],    RL_CH[l], C);
  }

  // --- global features on x (8 batches, N=2000) ---
  patch_feat_run(stream, net, nh, net.x, 8, 2000, w.gfeat, w);

  // --- patch features (64 batches of M=512, chunked 8 at a time) ---
  {
    long tot = 8L * 8 * 512 * 3;
    k_build_px<<<(unsigned)cdiv(tot, 256), 256, 0, stream>>>(
        net.x, net.patch_idx, w.px_all, 8, 8, 2000, 512, tot);
  }
  for (int c = 0; c < 8; ++c) {
    const float* pts = w.px_all + (long)c * 8 * 512 * 3;
    patch_feat_run(stream, net, nh, pts, 8, 512, w.pf + (long)c * 8 * 1024, w);
  }

  // --- relation map (raw reshape semantics of the reference) ---
  {
    long tot = 8L * 4096 * 64;
    k_build_rx<<<(unsigned)cdiv(tot, 256), 256, 0, stream>>>(w.gfeat, w.pf, w.rx, tot);
  }

  // --- three heads; output tuple order (rx_o, tx_o, ax_o) ---
  float* out = (float*)d_out;
  run_head(stream, net.rl_r,    nh.rl_r,    net.fc_r,    w.rx, out + 0,  512, 4, w);
  run_head(stream, net.rl_t,    nh.rl_t,    net.fc_t,    w.rx, out + 32, 256, 3, w);
  run_head(stream, net.rl_axis, nh.rl_axis, net.fc_axis, w.rx, out + 56, 256, 4, w);
}
